// trainable_similarity_map_62715112457054
// MI455X (gfx1250) — matrix operand order repaired and checked by reading, not compiled, not run on hardware
//
#include <hip/hip_runtime.h>

// MI455X / gfx1250 similarity-map kernel.
// cost = sumS2 - 2*cross + sumT2 ; cross term via V_WMMA_F32_16X16X32_F16
// (K = kx*3+c packed: 27/32 lanes used, B zero-padded so A tail garbage * 0 = 0).
// Box-filter terms via VALU; vertical 9-tap combine via branchless ds_add_f32
// scatter (invalid dx lands in cross-row padding; invalid rows land in a trash slot).
// One wave32 per (image, point); wave-private LDS slices -> no inter-wave sync.

typedef __attribute__((ext_vector_type(16))) _Float16 v16h;
typedef __attribute__((ext_vector_type(8)))  float    v8f;

namespace {
constexpr int Himg = 256, Wimg = 256, Cch = 3;
constexpr int NUMPTS = 2048;
constexpr int PSIZE = 25;             // kernel_size_search
constexpr int KSIZE = 9;              // kernel_size_center
constexpr int SP    = PSIZE / 2;      // 12
constexpr int PADI  = KSIZE / 2 + SP; // 16
constexpr int RROWS = PSIZE + KSIZE - 1;  // 33
constexpr int RSTRIDE = 112;          // halfs per LDS image row (>=105 used)
constexpr int CSTRIDE = 32;           // floats per cross row (25 used + 7 pad = dx dump)
constexpr int WAVES = 4;

constexpr unsigned IMG_HALFS = 3776;                  // 33*112=3696 + tail-tile slack
constexpr unsigned IMG_BYTES = IMG_HALFS * 2;         // 7552
constexpr unsigned SCR_BYTES = 4496;                  // max(rowsq 33*34*4, cross 25*32*4 + trash 128)
constexpr unsigned HS_BYTES  = RROWS * 32 * 4;        // 4224
constexpr unsigned WAVE_LDS  = IMG_BYTES + SCR_BYTES + HS_BYTES;  // 16272
constexpr unsigned BLOCK_LDS = WAVE_LDS * WAVES;      // 65088 <= 64KB
constexpr int RSQ_STRIDE = 34;
}

__global__ __launch_bounds__(WAVES * 32)
void simmap_wmma_kernel(const float* __restrict__ img0,
                        const float* __restrict__ img1,
                        const float* __restrict__ sigma_p,
                        const int*   __restrict__ coords,
                        float*       __restrict__ out)
{
  extern __shared__ char smem_raw[];
  const int lane = threadIdx.x & 31;
  const int wv   = threadIdx.x >> 5;
  const int task = blockIdx.x * WAVES + wv;       // 0..4095
  const int imgIdx = task >> 11;                  // 0: img, 1: img_sr
  const int n      = task & (NUMPTS - 1);

  char* base = smem_raw + (size_t)wv * WAVE_LDS;
  _Float16* L    = (_Float16*)base;                       // f16 window, (x,c)-interleaved
  float*    scr  = (float*)(base + IMG_BYTES);            // rowsq, later cross+trash
  float*    hsum = (float*)(base + IMG_BYTES + SCR_BYTES);

  const float* __restrict__ src = imgIdx ? img1 : img0;
  const int y0 = coords[2 * n];
  const int x0 = coords[2 * n + 1];

  // ---- Phase 1: stage 33x35x3 window, f32 -> f16, branchless zero padding ----
  for (int i = lane; i < RROWS * 35 * Cch; i += 32) {
    int r   = i / (35 * Cch);
    int rem = i - r * (35 * Cch);
    int xi  = rem / Cch;
    int c   = rem - xi * Cch;
    int y = y0 + r - PADI;
    int x = x0 + xi - PADI;
    bool inb = ((unsigned)y < (unsigned)Himg) & ((unsigned)x < (unsigned)Wimg);
    int yc = min(max(y, 0), Himg - 1);
    int xc = min(max(x, 0), Wimg - 1);
    float v = src[(c * Himg + yc) * Wimg + xc];   // always in-bounds load
    L[r * RSTRIDE + xi * Cch + c] = (_Float16)(inb ? v : 0.f);
  }
  __syncthreads();

  // ---- Phase 2: rowsq[r][x] = sum_c L^2 (f32) ----
  for (int i = lane; i < RROWS * 33; i += 32) {
    int r = i / 33, xi = i - r * 33;
    float s = 0.f;
#pragma unroll
    for (int c = 0; c < Cch; ++c) {
      float t = (float)L[r * RSTRIDE + xi * Cch + c];
      s += t * t;
    }
    scr[r * RSQ_STRIDE + xi] = s;
  }
  __syncthreads();

  // ---- Phase 3: horizontal 9-tap sum -> hsum[r][dx] ----
  for (int i = lane; i < RROWS * PSIZE; i += 32) {
    int r = i / PSIZE, dx = i - r * PSIZE;
    float s = 0.f;
#pragma unroll
    for (int kx = 0; kx < KSIZE; ++kx) s += scr[r * RSQ_STRIDE + dx + kx];
    hsum[r * 32 + dx] = s;
  }
  __syncthreads();

  // ---- Phase 4: repurpose scr as cross accumulator (+ trash row), zero it ----
  float* cross = scr;                 // 25*32 floats
  float* trash = scr + PSIZE * CSTRIDE;  // 32-float dump row (dead rowsq space)
  for (int i = lane; i < PSIZE * CSTRIDE + CSTRIDE; i += 32) cross[i] = 0.f;
  __syncthreads();

  // ---- Phase 5: B fragment = center patch weights, K = kx*3+c (27 used, rest 0) ----
  // 16-bit B 32x16 layout: lane l -> column N = l%16; lane half selects K base 0/8; elems 0..7 -> K, 8..15 -> K+16.
  const int ncol = lane & 15;                 // = ky
  const int kbB  = (lane < 16) ? 0 : 16;
  v16h bfrag;
#pragma unroll
  for (int e = 0; e < 16; ++e) {
    int k = (kbB >> 1) + (e < 8 ? e : e + 8);
    _Float16 v = (_Float16)0.f;
    if (ncol < KSIZE && k < Cch * KSIZE)
      v = L[(SP + ncol) * RSTRIDE + SP * Cch + k];
    bfrag[e] = v;
  }

  // ---- Phase 6: per-row WMMA correlation, software-pipelined, branchless scatter ----
  // 16-bit A 16x32 layout: lane half selects K base 0/8; elems 0..7 -> K, 8..15 -> K+16.
  const int half8 = (lane < 16) ? 0 : 8;
  const int mrow  = lane & 15;

  auto loadA = [&](int r, int mt) {
    const int arow = r * RSTRIDE + (mt * 16 + mrow) * Cch;
    v16h a;
#pragma unroll
    for (int e = 0; e < 16; ++e) {
      int K = half8 + (e < 8 ? e : e + 8);
      a[e] = L[arow + K];     // contiguous runs -> ds_load_b128 x2; K>=27 garbage * B=0
    }
    return a;
  };

  v16h a0 = loadA(0, 0);
  v16h a1 = loadA(0, 1);
  for (int r = 0; r < RROWS; ++r) {
    v8f z = {};
    v8f d0 = __builtin_amdgcn_wmma_f32_16x16x32_f16(false, a0, false, bfrag,
                                                    (short)0, z, false, false);
    v8f d1 = __builtin_amdgcn_wmma_f32_16x16x32_f16(false, a1, false, bfrag,
                                                    (short)0, z, false, false);
    // preload next row (clamped; redundant on last iter, keeps EXEC full, no branch)
    int rn = (r + 1 < RROWS) ? r + 1 : r;
    a0 = loadA(rn, 0);
    a1 = loadA(rn, 1);

    // D layout: lane l, VGPR vv -> M = vv + 8*(l>=16) (+16 for tile 1), N = l%16 = ky.
    // Row predicate is vv-invariant: one pointer select; dx>=25 lands in row padding.
    int dy = r - ncol;
    bool rowok = (ncol < KSIZE) & (dy >= 0) & (dy < PSIZE);
    float* rowp = rowok ? (cross + dy * CSTRIDE) : trash;
#pragma unroll
    for (int vv = 0; vv < 8; ++vv)
      atomicAdd(rowp + (half8 + vv), d0[vv]);          // tile 0: dx in [0,16)
#pragma unroll
    for (int vv = 0; vv < 8; ++vv)
      atomicAdd(rowp + (16 + half8 + vv), d1[vv]);     // tile 1: dx in [16,32)
  }
  __syncthreads();

  // ---- Phase 7: assemble cost, exp, store ----
  const float sig = fmaxf(sigma_p[0], 0.f);
  const float scale = -1.0f / ((float)(Cch * KSIZE * KSIZE) * sig);
  float ssc = 0.f;
#pragma unroll
  for (int ky = 0; ky < KSIZE; ++ky) ssc += hsum[(SP + ky) * 32 + SP];

  float* optr = out + ((size_t)imgIdx * NUMPTS + n) * (PSIZE * PSIZE);
  for (int i = lane; i < PSIZE * PSIZE; i += 32) {
    int dy = i / PSIZE, dx = i - dy * PSIZE;
    float ss = 0.f;
#pragma unroll
    for (int ky = 0; ky < KSIZE; ++ky) ss += hsum[(dy + ky) * 32 + dx];
    float cost = ss - 2.f * cross[dy * CSTRIDE + dx] + ssc;
    optr[i] = __expf(cost * scale + 1e-20f);
  }
}

extern "C" void kernel_launch(void* const* d_in, const int* in_sizes, int n_in,
                              void* d_out, int out_size, void* d_ws, size_t ws_size,
                              hipStream_t stream) {
  const float* img0   = (const float*)d_in[0];
  const float* img1   = (const float*)d_in[1];
  const float* sigma  = (const float*)d_in[2];
  const int*   coords = (const int*)d_in[3];
  float* out = (float*)d_out;

  dim3 grid((2 * NUMPTS) / WAVES);   // 1024 blocks
  dim3 block(WAVES * 32);            // 4 waves of 32
  hipLaunchKernelGGL(simmap_wmma_kernel, grid, block, BLOCK_LDS, stream,
                     img0, img1, sigma, coords, out);
}
